// HAFT_9225589751830
// MI455X (gfx1250) — matrix operand order repaired and checked by reading, not compiled, not run on hardware
//
#include <hip/hip_runtime.h>
#include <hip/hip_bf16.h>
#include <math.h>

// ---------------------------------------------------------------------------
// HAFT forward for gfx1250 (MI455X).  All dense contractions are WMMA:
//   * 2D DFT / inverse DFT  -> batched real GEMMs on V_WMMA_F32_16X16X4_F32
//   * FilterPredictor MLPs  -> V_WMMA_F32_16X16X32_F16, weights staged to LDS
//     with GLOBAL_LOAD_ASYNC_TO_LDS_B32 (+ s_wait_asynccnt)
// Intermediates live in a (B*C, 128, 128) "canvas" layout so every level's
// patches are addressed in place with strides (no patch extraction copies).
// ---------------------------------------------------------------------------

typedef __attribute__((ext_vector_type(2)))  float    v2f;
typedef __attribute__((ext_vector_type(8)))  float    v8f;
typedef __attribute__((ext_vector_type(16))) _Float16 v16h;

__device__ __forceinline__ v8f wmma_k4(v2f a, v2f b, v8f c) {
  // D = A(16x4 f32) * B(4x16 f32) + C(16x16 f32)
  return __builtin_amdgcn_wmma_f32_16x16x4_f32(false, a, false, b, (short)0, c,
                                               false, false);
}
__device__ __forceinline__ v8f wmma_h(v16h a, v16h b, v8f c) {
  // D = A(16x32 f16) * B(32x16 f16) + C(16x16 f32)
  return __builtin_amdgcn_wmma_f32_16x16x32_f16(false, a, false, b, (short)0, c,
                                                false, false);
}

// Async copy of one dword from global to LDS (ASYNCcnt-tracked, no VGPR).
__device__ __forceinline__ void async_g2l_b32(void* lds_dst, const void* gsrc) {
  unsigned int loff = (unsigned int)(unsigned long long)lds_dst; // LDS offset
  asm volatile("global_load_async_to_lds_b32 %0, %1, off"
               :: "v"(loff), "v"(gsrc) : "memory");
}
__device__ __forceinline__ void async_wait0() {
  asm volatile("s_wait_asynccnt 0x0" ::: "memory");
}

#define BC      256     // B*C
#define CANV    16384   // 128*128
#define ROWS    4096    // BC * 16 deepest patches
#define HID     128
#define BINS    16

// ---------------------------------------------------------------------------
// Twiddle tables (cos/sin of 2*pi*j*k/N) for N=128,64,32 + radial bin indices.
// ---------------------------------------------------------------------------
__global__ void haft_tables_kernel(float* tc128, float* ts128,
                                   float* tc64,  float* ts64,
                                   float* tc32,  float* ts32, int* ridx) {
  const float TWO_PI = 6.28318530717958647692f;
  int t = blockIdx.x * blockDim.x + threadIdx.x;
  if (t < 128 * 128) {
    int j = t >> 7, k = t & 127;
    float a = TWO_PI * (float)((j * k) & 127) * (1.0f / 128.0f);
    float s, c; __sincosf(a, &s, &c);
    tc128[t] = c; ts128[t] = s;
  }
  if (t < 64 * 64) {
    int j = t >> 6, k = t & 63;
    float a = TWO_PI * (float)((j * k) & 63) * (1.0f / 64.0f);
    float s, c; __sincosf(a, &s, &c);
    tc64[t] = c; ts64[t] = s;
  }
  if (t < 32 * 32) {
    int j = t >> 5, k = t & 31;
    float a = TWO_PI * (float)((j * k) & 31) * (1.0f / 32.0f);
    float s, c; __sincosf(a, &s, &c);
    tc32[t] = c; ts32[t] = s;
    // radial Chebyshev bins for the deepest (32x32) level
    float d  = fmaxf(fabsf((float)j - 16.0f), fabsf((float)k - 16.0f));
    float dn = d * (1.0f / 16.0f) * (float)(BINS - 1);
    int id = (int)floorf(dn);
    id = id < 0 ? 0 : (id > BINS - 1 ? BINS - 1 : id);
    ridx[t] = id;
  }
}

// f32 -> f16 weight conversion into WMMA-B pair-packed layout:
//   dst[((k>>1)*N + n)*2 + (k&1)] = (f16) src[k*N + n]
// so the two K-halves a lane needs are one aligned dword.
__global__ void haft_cvt_pack_kernel(const float* __restrict__ src,
                                     _Float16* __restrict__ dst,
                                     int K, int N) {
  int t = blockIdx.x * blockDim.x + threadIdx.x;
  if (t >= K * N) return;
  int k = t / N, n = t - k * N;
  dst[(size_t)(((k >> 1) * N + n) << 1) + (k & 1)] = (_Float16)src[t];
}

// ---------------------------------------------------------------------------
// Generic complex DFT pass as batched WMMA-f32 GEMM.
//   OUTr = C*INr + sgn1 * S*INi        (INi may be null -> treated as 0)
//   OUTi = sgn2 * S*INr + C*INi
// One 16x16 output tile per wave; K stepped by 4 (V_WMMA_F32_16X16X4_F32).
// mode 0: write (re,im); mode 1: write (mag,phase); mode 2: write re*scale.
// Per-patch base = bc*CANV + pi*N*128 + pj*N; elements addressed by strides.
// ---------------------------------------------------------------------------
__global__ void haft_dft_pass_kernel(const float* __restrict__ twc,
                                     const float* __restrict__ tws,
                                     const float* __restrict__ in_r,
                                     const float* __restrict__ in_i,
                                     float* __restrict__ o0,
                                     float* __restrict__ o1,
                                     int N, int nside,
                                     float sgn1, float sgn2, float scale,
                                     int mode, int in_sk, int in_sn,
                                     int out_sm, int out_sn) {
  const int patch = blockIdx.x;
  const int pp    = nside * nside;
  const int bc    = patch / pp;
  const int pin   = patch - bc * pp;
  const int pi    = pin / nside, pj = pin - pi * nside;
  const int base  = bc * CANV + pi * N * 128 + pj * N;

  const int tps   = N >> 4;                       // tiles per side
  const int tile  = blockIdx.y * 4 + (threadIdx.x >> 5);
  if (tile >= tps * tps) return;                  // wave-uniform exit
  const int trow = tile / tps, tcol = tile - trow * tps;

  const int lane = threadIdx.x & 31;
  const int l15  = lane & 15;
  const int hs   = lane >> 4;                     // lane half select
  const int mrow = trow * 16 + l15;               // A row for this lane
  const int ncol = tcol * 16 + l15;               // B col for this lane
  const bool hasI = (in_i != nullptr);

  v8f acc_cr = {}, acc_sr = {}, acc_ci = {}, acc_si = {};

  for (int kk = 0; kk < N; kk += 4) {
    // A (16x4 f32): lanes 0-15 hold K=kk+0/1, lanes 16-31 hold K=kk+2/3
    const int ka = kk + hs * 2;
    v2f ac, as;
    ac.x = twc[mrow * N + ka];  ac.y = twc[mrow * N + ka + 1];
    as.x = tws[mrow * N + ka];  as.y = tws[mrow * N + ka + 1];
    // B (4x16 f32): VGPR0 = rows {0,2}, VGPR1 = rows {1,3} across lane halves
    const int kb = kk + hs * 2;
    v2f br;
    br.x = in_r[base + kb * in_sk + ncol * in_sn];
    br.y = in_r[base + (kb + 1) * in_sk + ncol * in_sn];
    acc_cr = wmma_k4(ac, br, acc_cr);
    acc_sr = wmma_k4(as, br, acc_sr);
    if (hasI) {
      v2f bi;
      bi.x = in_i[base + kb * in_sk + ncol * in_sn];
      bi.y = in_i[base + (kb + 1) * in_sk + ncol * in_sn];
      acc_ci = wmma_k4(ac, bi, acc_ci);
      acc_si = wmma_k4(as, bi, acc_si);
    }
  }

  #pragma unroll
  for (int v = 0; v < 8; ++v) {
    const int m = trow * 16 + v + hs * 8;
    const int n = tcol * 16 + l15;
    const int off = base + m * out_sm + n * out_sn;
    float re = acc_cr[v] + sgn1 * acc_si[v];
    float im = sgn2 * acc_sr[v] + acc_ci[v];
    if (mode == 0)      { o0[off] = re; o1[off] = im; }
    else if (mode == 1) { o0[off] = sqrtf(re * re + im * im);
                          o1[off] = atan2f(im, re); }
    else                { o0[off] = re * scale; }
  }
}

// ---------------------------------------------------------------------------
// Patch encoder: conv3x3(1->16)+BN+ReLU -> 4x4 block mean -> conv3x3(16->32)
// +BN+ReLU -> global mean -> linear(32->64).  One workgroup per patch.
// ---------------------------------------------------------------------------
__global__ void haft_encoder_kernel(const float* __restrict__ canvas,
                                    int N, int nside,
                                    const float* b1, const float* b2,
                                    const float* be1, const float* be2,
                                    const float* bl,
                                    const float* g1, const float* g2,
                                    const float* m1, const float* m2,
                                    const float* rv1, const float* rv2,
                                    const float* w1, const float* w2,
                                    const float* wl,
                                    float* __restrict__ ctx, int ctxOff) {
  __shared__ float pool[16 * 16];     // 16 channels x 4x4 pooled map
  __shared__ float h2m[32];
  __shared__ float sW1[144];
  __shared__ float sSc[16], sSh[16];

  const int patch = blockIdx.x;
  const int pp = nside * nside;
  const int bc = patch / pp;
  const int pin = patch - bc * pp;
  const int pi = pin / nside, pj = pin - pi * nside;
  const float* base = canvas + bc * CANV + pi * N * 128 + pj * N;
  const int tid = threadIdx.x;

  if (tid < 144) sW1[tid] = w1[tid];
  if (tid < 16) {
    float inv = rsqrtf(rv1[tid] + 1e-5f);
    sSc[tid] = g1[tid] * inv;
    sSh[tid] = be1[tid] - m1[tid] * sSc[tid];
  }
  if (tid < 256) pool[tid] = 0.0f;
  __syncthreads();

  const int cell = N >> 2;            // side of each 4x4 pooling block
  const float cInv = 1.0f / (float)(cell * cell);
  for (int pix = tid; pix < N * N; pix += 256) {
    const int r = pix / N, c = pix - r * N;
    float x[9];
    #pragma unroll
    for (int dy = 0; dy < 3; ++dy)
      #pragma unroll
      for (int dx = 0; dx < 3; ++dx) {
        int rr = r + dy - 1, cc = c + dx - 1;
        x[dy * 3 + dx] = (rr >= 0 && rr < N && cc >= 0 && cc < N)
                             ? base[rr * 128 + cc] : 0.0f;
      }
    const int pc = (r / cell) * 4 + (c / cell);
    #pragma unroll
    for (int oc = 0; oc < 16; ++oc) {
      float acc = b1[oc];
      #pragma unroll
      for (int k = 0; k < 9; ++k) acc += sW1[oc * 9 + k] * x[k];
      acc = fmaxf(acc * sSc[oc] + sSh[oc], 0.0f);
      atomicAdd(&pool[oc * 16 + pc], acc * cInv);
    }
  }
  __syncthreads();

  if (tid < 32) {
    const float inv = rsqrtf(rv2[tid] + 1e-5f);
    const float sc = g2[tid] * inv;
    const float sh = be2[tid] - m2[tid] * sc;
    float mean = 0.0f;
    for (int py = 0; py < 4; ++py)
      for (int px = 0; px < 4; ++px) {
        float acc = b2[tid];
        for (int ic = 0; ic < 16; ++ic)
          #pragma unroll
          for (int ky = 0; ky < 3; ++ky)
            #pragma unroll
            for (int kx = 0; kx < 3; ++kx) {
              int yy = py + ky - 1, xx = px + kx - 1;
              if (yy >= 0 && yy < 4 && xx >= 0 && xx < 4)
                acc += w2[((tid * 16 + ic) * 3 + ky) * 3 + kx] *
                       pool[ic * 16 + yy * 4 + xx];
            }
        mean += fmaxf(acc * sc + sh, 0.0f);
      }
    h2m[tid] = mean * (1.0f / 16.0f);
  }
  __syncthreads();

  if (tid < 64) {
    float acc = bl[tid];
    #pragma unroll
    for (int i = 0; i < 32; ++i) acc += h2m[i] * wl[i * 64 + tid];
    ctx[(size_t)patch * 128 + ctxOff + tid] = acc;
  }
}

// ---------------------------------------------------------------------------
// Build fused context matrix (ROWS x 384) in f16 with level embeddings.
// ---------------------------------------------------------------------------
__global__ void haft_fuse_kernel(const float* __restrict__ ctx0,
                                 const float* __restrict__ ctx1,
                                 const float* __restrict__ ctx2,
                                 const float* __restrict__ lp_w,
                                 const float* __restrict__ lp_b,
                                 _Float16* __restrict__ fused) {
  int t = blockIdx.x * blockDim.x + threadIdx.x;
  if (t >= ROWS * 384) return;
  int row = t / 384, col = t - row * 384;
  int l = col >> 7, j = col & 127;
  int bc = row >> 4, p = row & 15;
  int mapped, P;
  if (l == 0)      { mapped = 0; P = 1; }
  else if (l == 1) { mapped = ((p >> 3) << 1) | ((p & 3) >> 1); P = 4; }
  else             { mapped = p; P = 16; }
  const float* ctx = (l == 0) ? ctx0 : (l == 1) ? ctx1 : ctx2;
  float v = ctx[(size_t)(bc * P + mapped) * 128 + j] +
            ((float)l * lp_w[j] + lp_b[j]);
  fused[t] = (_Float16)v;
}

// ---------------------------------------------------------------------------
// f16 GEMM + bias + optional ReLU via V_WMMA_F32_16X16X32_F16.
// Weights (pair-packed) are staged into LDS with async global->LDS copies.
// One 16x16 tile per wave; grid = (M/16, ceil(N/64)), block = 128 (4 waves).
// ---------------------------------------------------------------------------
__global__ void haft_gemm_f16_kernel(const _Float16* __restrict__ A, int lda,
                                     int aOff,
                                     const _Float16* __restrict__ Wp,
                                     const float* __restrict__ bias,
                                     float* __restrict__ outF,
                                     _Float16* __restrict__ outH,
                                     int M, int N, int K, int act) {
  __shared__ unsigned int sWd[96 * 64];         // (K/2) x cols dwords, <=24 KB
  const int tid  = threadIdx.x;
  const int lane = tid & 31;
  const int wv   = tid >> 5;
  const int l15  = lane & 15, hs = lane >> 4;

  // ---- async-stage this block's weight stripe (cols [n0, n0+cols)) ----
  const int n0   = blockIdx.y * 64;
  const int cols = (N - n0 < 64) ? (N - n0) : 64;
  const int ndw  = (K >> 1) * cols;
  const unsigned int* g = (const unsigned int*)Wp;
  for (int d = tid; d < ndw; d += 128) {
    int k2 = d / cols, nn = d - k2 * cols;
    async_g2l_b32(&sWd[d], &g[k2 * N + n0 + nn]);
  }
  async_wait0();
  __syncthreads();

  const int nt = blockIdx.y * 4 + wv;
  if (nt * 16 >= N) return;                     // wave-uniform exit
  const int mt = blockIdx.x;
  const int lcol = wv * 16 + l15;
  const _Float16* sWh = (const _Float16*)sWd;
  const _Float16* Ar = A + (size_t)(mt * 16) * lda + aOff + (size_t)l15 * lda;

  v8f acc = {};
  for (int kk = 0; kk < K; kk += 32) {
    v16h af, bf;
    // A 16x32 f16: lanes0-15 rows M=l15 K = 0..15; +16 for hi lanes
    #pragma unroll
    for (int v = 0; v < 8; ++v) {
      int k = kk + 2 * v + hs * 16;
      af[2 * v]     = Ar[k];
      af[2 * v + 1] = Ar[k + 1];
    }
    // B 32x16 f16: VGPR v holds K = 2v,2v+1 (+16 for hi lanes); one LDS dword
    #pragma unroll
    for (int v = 0; v < 8; ++v) {
      int k  = kk + 2 * v + hs * 16;
      int dw = (k >> 1) * cols + lcol;
      bf[2 * v]     = sWh[2 * dw];
      bf[2 * v + 1] = sWh[2 * dw + 1];
    }
    if (kk + 32 < K)
      __builtin_prefetch(&Ar[kk + 32], 0, 3);
    acc = wmma_h(af, bf, acc);
  }

  const int nB = nt * 16 + l15;
  const float bv = bias ? bias[nB] : 0.0f;
  #pragma unroll
  for (int v = 0; v < 8; ++v) {
    const int m = mt * 16 + v + hs * 8;
    float r = acc[v] + bv;
    if (act) r = fmaxf(r, 0.0f);
    if (outF) outF[(size_t)m * N + nB] = r;
    if (outH) outH[(size_t)m * N + nB] = (_Float16)r;
  }
}

// Softmax(16) on mag head, tanh on phase head; one row per thread.
__global__ void haft_heads_kernel(float* __restrict__ wmag,
                                  float* __restrict__ wphase) {
  int row = blockIdx.x * blockDim.x + threadIdx.x;
  if (row >= ROWS) return;
  float* a = wmag + (size_t)row * BINS;
  float mx = a[0];
  #pragma unroll
  for (int i = 1; i < BINS; ++i) mx = fmaxf(mx, a[i]);
  float s = 0.0f;
  #pragma unroll
  for (int i = 0; i < BINS; ++i) { float e = __expf(a[i] - mx); a[i] = e; s += e; }
  float inv = 1.0f / s;
  #pragma unroll
  for (int i = 0; i < BINS; ++i) a[i] *= inv;
  float* b = wphase + (size_t)row * BINS;
  #pragma unroll
  for (int i = 0; i < BINS; ++i) b[i] = tanhf(b[i]);
}

// Apply radial filters to deepest-level spectrum, rebuild complex Z.
__global__ void haft_filter_kernel(const float* __restrict__ magD,
                                   const float* __restrict__ phaseD,
                                   const float* __restrict__ wmag,
                                   const float* __restrict__ wphase,
                                   const int* __restrict__ ridx,
                                   float* __restrict__ Zr,
                                   float* __restrict__ Zi) {
  int t = blockIdx.x * blockDim.x + threadIdx.x;
  if (t >= BC * CANV) return;
  int bc = t >> 14, pos = t & 16383;
  int r = pos >> 7, c = pos & 127;
  int p = ((r >> 5) << 2) | (c >> 5);        // deepest patch index (4x4 grid)
  int d = ridx[((r & 31) << 5) | (c & 31)];
  size_t w = ((size_t)((bc << 4) + p)) * BINS + d;
  float m  = magD[t] * wmag[w];
  float ph = phaseD[t] + wphase[w];
  float s, cs; __sincosf(ph, &s, &cs);
  Zr[t] = m * cs;
  Zi[t] = m * s;
}

// 1x1 output projection: out[b,o,h,w] = sum_c rec[b,c,h,w]*W[o,c] + b[o].
__global__ void haft_proj_kernel(const float* __restrict__ rec,
                                 const float* __restrict__ pw,
                                 const float* __restrict__ pb,
                                 float* __restrict__ out) {
  __shared__ float sw[1024];
  __shared__ float sb[32];
  int tid = threadIdx.x;
  for (int i = tid; i < 1024; i += 256)
    async_g2l_b32(&sw[i], &pw[i]);             // async weight broadcast
  if (tid < 32) sb[tid] = pb[tid];
  async_wait0();
  __syncthreads();
  int pix = blockIdx.x * 256 + tid;            // over B*H*W = 131072
  if (pix >= 8 * CANV) return;
  int b = pix >> 14, hw = pix & 16383;
  float x[32];
  #pragma unroll
  for (int c = 0; c < 32; ++c) x[c] = rec[(((size_t)b * 32 + c) << 14) + hw];
  #pragma unroll
  for (int o = 0; o < 32; ++o) {
    float acc = sb[o];
    #pragma unroll
    for (int c = 0; c < 32; ++c) acc += sw[o * 32 + c] * x[c];
    out[(((size_t)b * 32 + o) << 14) + hw] = acc;
  }
}

// ---------------------------------------------------------------------------
// Host-side orchestration.
// d_in layout (jax pytree alphabetical flatten):
//   [0] F_in
//   [1 + (lvl*2+mp)*14 + k] encoder tensors, k in
//       {b1,b2,be1,be2,bl,g1,g2,m1,m2,v1,v2,w1,w2,wl}
//   [85] lp_b  [86] lp_w
//   [87..92]  mag_mlp   {b1,b2,b3,w1,w2,w3}
//   [93..98]  phase_mlp {b1,b2,b3,w1,w2,w3}
//   [99] proj_b  [100] proj_w
// ---------------------------------------------------------------------------
extern "C" void kernel_launch(void* const* d_in, const int* in_sizes, int n_in,
                              void* d_out, int out_size, void* d_ws,
                              size_t ws_size, hipStream_t stream) {
  (void)in_sizes; (void)n_in; (void)out_size; (void)ws_size;
  const float* F_in = (const float*)d_in[0];
  auto enc = [&](int lvl, int mp, int k) -> const float* {
    return (const float*)d_in[1 + (lvl * 2 + mp) * 14 + k];
  };
  const float* lp_b = (const float*)d_in[85];
  const float* lp_w = (const float*)d_in[86];
  const float* mlpP[2][6];
  for (int i = 0; i < 6; ++i) { mlpP[0][i] = (const float*)d_in[87 + i];
                                mlpP[1][i] = (const float*)d_in[93 + i]; }
  const float* proj_b = (const float*)d_in[99];
  const float* proj_w = (const float*)d_in[100];

  // ---- workspace carve-up (floats) ----
  float* W = (float*)d_ws;
  size_t o = 0;
  auto take = [&](size_t n) { float* p = W + o; o += n; return p; };
  float* tc128 = take(16384); float* ts128 = take(16384);
  float* tc64  = take(4096);  float* ts64  = take(4096);
  float* tc32  = take(1024);  float* ts32  = take(1024);
  int*   ridx  = (int*)take(1024);
  float* magA   = take((size_t)BC * CANV);   // scratch mag canvas (lvl 0/1)
  float* phaseA = take((size_t)BC * CANV);
  float* magD   = take((size_t)BC * CANV);   // deepest-level mag/phase
  float* phaseD = take((size_t)BC * CANV);
  float* Tr     = take((size_t)BC * CANV);   // DFT row-pass intermediate
  float* Ti     = take((size_t)BC * CANV);
  float* Zr     = take((size_t)BC * CANV);   // filtered complex spectrum
  float* Zi     = take((size_t)BC * CANV);
  float* ctx0 = take((size_t)BC * 1 * 128);
  float* ctx1 = take((size_t)BC * 4 * 128);
  float* ctx2 = take((size_t)BC * 16 * 128);
  float* wmag   = take((size_t)ROWS * BINS);
  float* wphase = take((size_t)ROWS * BINS);
  float* rec    = take((size_t)BC * CANV);   // (B*C,128,128)
  _Float16* fused = (_Float16*)take((size_t)ROWS * 384 / 2);
  _Float16* h1    = (_Float16*)take((size_t)ROWS * HID / 2);
  _Float16* h2    = (_Float16*)take((size_t)ROWS * HID / 2);
  _Float16* wh[2][3];
  int wK[3] = {192, HID, HID};
  int wN[3] = {HID, HID, BINS};
  for (int mp = 0; mp < 2; ++mp)
    for (int l = 0; l < 3; ++l)
      wh[mp][l] = (_Float16*)take((size_t)wK[l] * wN[l] / 2);

  // ---- tables + weight conversion (pair-packed f16) ----
  haft_tables_kernel<<<64, 256, 0, stream>>>(tc128, ts128, tc64, ts64,
                                             tc32, ts32, ridx);
  for (int mp = 0; mp < 2; ++mp)
    for (int l = 0; l < 3; ++l) {
      int n = wK[l] * wN[l];
      haft_cvt_pack_kernel<<<(n + 255) / 256, 256, 0, stream>>>(
          mlpP[mp][3 + l], wh[mp][l], wK[l], wN[l]);
    }

  // ---- per-level forward DFT + encoders ----
  struct Lvl { int N, n; const float *tc, *ts; float *mg, *ph; float* ctx; };
  Lvl L[3] = {{128, 1, tc128, ts128, magA, phaseA, ctx0},
              {64,  2, tc64,  ts64,  magA, phaseA, ctx1},
              {32,  4, tc32,  ts32,  magD, phaseD, ctx2}};
  for (int l = 0; l < 3; ++l) {
    const int patches = BC * L[l].n * L[l].n;
    const int tiles = (L[l].N / 16) * (L[l].N / 16);
    dim3 grid(patches, (tiles + 3) / 4);
    // pass 1: T = (C - iS) * X          (X real, read in place from F_in)
    haft_dft_pass_kernel<<<grid, 128, 0, stream>>>(
        L[l].tc, L[l].ts, F_in, nullptr, Tr, Ti,
        L[l].N, L[l].n, 0.0f, -1.0f, 1.0f, /*mode=*/0,
        /*in_sk=*/128, /*in_sn=*/1, /*out_sm=*/128, /*out_sn=*/1);
    // pass 2: Y = T * (C - iS)  -> mag/phase (transposed read, transposed write)
    haft_dft_pass_kernel<<<grid, 128, 0, stream>>>(
        L[l].tc, L[l].ts, Tr, Ti, L[l].mg, L[l].ph,
        L[l].N, L[l].n, +1.0f, -1.0f, 1.0f, /*mode=*/1,
        /*in_sk=*/1, /*in_sn=*/128, /*out_sm=*/1, /*out_sn=*/128);
    for (int mp = 0; mp < 2; ++mp) {
      const float* cv = (mp == 0) ? L[l].mg : L[l].ph;
      haft_encoder_kernel<<<patches, 256, 0, stream>>>(
          cv, L[l].N, L[l].n,
          enc(l, mp, 0), enc(l, mp, 1), enc(l, mp, 2), enc(l, mp, 3),
          enc(l, mp, 4), enc(l, mp, 5), enc(l, mp, 6), enc(l, mp, 7),
          enc(l, mp, 8), enc(l, mp, 9), enc(l, mp, 10), enc(l, mp, 11),
          enc(l, mp, 12), enc(l, mp, 13), L[l].ctx, mp * 64);
    }
  }

  // ---- fuse contexts + two 3-layer MLPs (WMMA f16, async-staged weights) ----
  haft_fuse_kernel<<<(ROWS * 384 + 255) / 256, 256, 0, stream>>>(
      ctx0, ctx1, ctx2, lp_w, lp_b, fused);
  for (int mp = 0; mp < 2; ++mp) {
    float* head = (mp == 0) ? wmag : wphase;
    haft_gemm_f16_kernel<<<dim3(ROWS / 16, 2), 128, 0, stream>>>(
        fused, 384, mp * 192, wh[mp][0], mlpP[mp][0], nullptr, h1,
        ROWS, HID, 192, /*act=*/1);
    haft_gemm_f16_kernel<<<dim3(ROWS / 16, 2), 128, 0, stream>>>(
        h1, HID, 0, wh[mp][1], mlpP[mp][1], nullptr, h2,
        ROWS, HID, HID, /*act=*/1);
    haft_gemm_f16_kernel<<<dim3(ROWS / 16, 1), 128, 0, stream>>>(
        h2, HID, 0, wh[mp][2], mlpP[mp][2], head, nullptr,
        ROWS, BINS, HID, /*act=*/0);
  }
  haft_heads_kernel<<<ROWS / 256, 256, 0, stream>>>(wmag, wphase);

  // ---- radial filter, inverse DFT (WMMA f32), projection ----
  haft_filter_kernel<<<(BC * CANV) / 256, 256, 0, stream>>>(
      magD, phaseD, wmag, wphase, ridx, Zr, Zi);
  {
    dim3 grid(BC * 16, 1);   // 4096 patches of 32x32; 4 tiles -> 1 y-group
    // inverse pass 1: T = (C + iS) * Z
    haft_dft_pass_kernel<<<grid, 128, 0, stream>>>(
        tc32, ts32, Zr, Zi, Tr, Ti, 32, 4, -1.0f, +1.0f, 1.0f, /*mode=*/0,
        128, 1, 128, 1);
    // inverse pass 2: rec = Re{ T * (C + iS) } / (32*32)
    haft_dft_pass_kernel<<<grid, 128, 0, stream>>>(
        tc32, ts32, Tr, Ti, rec, nullptr, 32, 4, -1.0f, +1.0f,
        1.0f / 1024.0f, /*mode=*/2, 1, 128, 1, 128);
  }
  haft_proj_kernel<<<512, 256, 0, stream>>>(rec, proj_w, proj_b,
                                            (float*)d_out);
}
